// FlashAttention_23416161698398
// MI455X (gfx1250) — matrix operand order repaired and checked by reading, not compiled, not run on hardware
//
#include <hip/hip_runtime.h>

// ---------------------------------------------------------------------------
// Types
// ---------------------------------------------------------------------------
typedef __attribute__((ext_vector_type(16))) __bf16 v16bf;
typedef __attribute__((ext_vector_type(4)))  __bf16 v4bf;
typedef __attribute__((ext_vector_type(8)))  float  v8f;
typedef __attribute__((ext_vector_type(4)))  int    v4i;

// Pointer flavors for the async-copy builtin (param 1 is a global-AS pointer
// to a 4 x i32 vector per the compiler diagnostic; LDS side mirrors it).
typedef __attribute__((address_space(1))) v4i* g_v4i_ptr;
typedef __attribute__((address_space(3))) v4i* l_v4i_ptr;

// Problem constants (match reference)
static constexpr int Bn = 2;
static constexpr int Tn = 2048;
static constexpr int Dm = 1024;
static constexpr int Hn = 16;
static constexpr int DH = 64;

// ---------------------------------------------------------------------------
// Async global->LDS path (gfx1250). Guarded so absence degrades gracefully.
// ---------------------------------------------------------------------------
#if __has_builtin(__builtin_amdgcn_global_load_async_to_lds_b128)
#define USE_ASYNC 1
#else
#define USE_ASYNC 0
#endif

__device__ __forceinline__ void copy16_g2l(const __bf16* g, __bf16* l) {
#if USE_ASYNC
  __builtin_amdgcn_global_load_async_to_lds_b128((g_v4i_ptr)g, (l_v4i_ptr)l, 0,
                                                 0);
#else
  *(int4*)l = *(const int4*)g;
#endif
}

__device__ __forceinline__ void wait_fills() {
#if USE_ASYNC
#if __has_builtin(__builtin_amdgcn_s_wait_asynccnt)
  __builtin_amdgcn_s_wait_asynccnt(0);
#else
  asm volatile("s_wait_asynccnt 0" ::: "memory");
#endif
#endif
}

// ---------------------------------------------------------------------------
// WMMA helpers (CDNA5 wave32, V_WMMA_F32_16X16X32_BF16)
// ---------------------------------------------------------------------------
__device__ __forceinline__ v8f wmma_bf16(v16bf a, v16bf b, v8f c) {
  return __builtin_amdgcn_wmma_f32_16x16x32_bf16(
      false, a, false, b, (short)0, c, false, false);
}

// A-matrix fragment: 16x32 bf16, MxK (ISA layout); LDS [m][k], pairs merge to
// ds_load_b128.
__device__ __forceinline__ v16bf frag_a_16x32(const __bf16* p, int strideM) {
  const int lane = threadIdx.x & 31;
  const int m = lane & 15;
  const int hi = lane >> 4;
  v16bf f;
#pragma unroll
  for (int vg = 0; vg < 8; ++vg) {
    const int kb = 2 * (vg & 3) + ((vg >= 4) ? 16 : 0) + (hi ? 8 : 0);
    f[2 * vg]     = p[m * strideM + kb];
    f[2 * vg + 1] = p[m * strideM + kb + 1];
  }
  return f;
}

// B-matrix fragment: 32x16 bf16, KxN, read from LDS stored as [N][K].
__device__ __forceinline__ v16bf frag_b_32x16(const __bf16* p, int strideN) {
  const int lane = threadIdx.x & 31;
  const int n = lane & 15;
  const int hi = lane >> 4;
  v16bf f;
#pragma unroll
  for (int vg = 0; vg < 8; ++vg) {
    const int kb = 2 * (vg & 3) + ((vg >= 4) ? 16 : 0) + (hi ? 8 : 0);
    f[2 * vg]     = p[n * strideN + kb];
    f[2 * vg + 1] = p[n * strideN + kb + 1];
  }
  return f;
}

__device__ __forceinline__ float half_reduce_max(float v) {
#pragma unroll
  for (int off = 1; off < 16; off <<= 1) v = fmaxf(v, __shfl_xor(v, off, 32));
  return v;
}
__device__ __forceinline__ float half_reduce_sum(float v) {
#pragma unroll
  for (int off = 1; off < 16; off <<= 1) v += __shfl_xor(v, off, 32);
  return v;
}

// ---------------------------------------------------------------------------
// Precision staging kernels: f32 -> bf16 copy, and f32 [K][N] -> bf16 [N][K]
// transpose (gives the exact LDS layout the B fragment wants, enabling pure
// byte-copy async tile fills in the GEMMs).
// ---------------------------------------------------------------------------
__global__ __launch_bounds__(256) void cvt_f32_bf16_kernel(
    const float* __restrict__ src, __bf16* __restrict__ dst, int n4) {
  const int i = blockIdx.x * 256 + threadIdx.x;
  if (i >= n4) return;
  const float4 v = ((const float4*)src)[i];
  v4bf o;
  o[0] = (__bf16)v.x; o[1] = (__bf16)v.y; o[2] = (__bf16)v.z; o[3] = (__bf16)v.w;
  ((v4bf*)dst)[i] = o;
}

__global__ __launch_bounds__(256) void transpose_f32_bf16_kernel(
    const float* __restrict__ W, __bf16* __restrict__ Wt, int K, int N) {
  __shared__ float tile[32][33];
  const int tx = threadIdx.x & 31, ty = threadIdx.x >> 5;  // 32 x 8
  const int n0 = blockIdx.x * 32, k0 = blockIdx.y * 32;
#pragma unroll
  for (int j = 0; j < 4; ++j)
    tile[ty + j * 8][tx] = W[(long)(k0 + ty + j * 8) * N + n0 + tx];
  __syncthreads();
#pragma unroll
  for (int j = 0; j < 4; ++j)
    Wt[(long)(n0 + ty + j * 8) * K + k0 + tx] = (__bf16)tile[tx][ty + j * 8];
}

// ---------------------------------------------------------------------------
// GEMM: C[M,N] = Ab[M,K](bf16) * Bt[N,K](bf16, pre-transposed) + bias[N]
//   EPI==0 : store f32 into Cout
//   EPI==1 : QKV epilogue -> scatter bf16 Q(*1/8)/K/V into [B,H,T,d]
// 256 threads = 8 waves (4 M x 2 N); block tile 128x128, BK=32, each wave owns
// 32x64 (2 A-frags x 4 B-frags -> 8 WMMA / K-step). Double-buffered async
// LDS fills.
// ---------------------------------------------------------------------------
template <int EPI>
__global__ __launch_bounds__(256) void gemm_bf16_kernel(
    const __bf16* __restrict__ Ab, const __bf16* __restrict__ Bt,
    const float* __restrict__ bias, float* __restrict__ Cout,
    __bf16* __restrict__ Qb, __bf16* __restrict__ Kb, __bf16* __restrict__ Vb,
    int M, int N, int K) {
  constexpr int BM = 128, BN = 128, BK = 32;
  __shared__ __bf16 lA[2][BM * BK];  // [m][k]
  __shared__ __bf16 lB[2][BN * BK];  // [n][k]

  const int tid = threadIdx.x;
  const int lane = tid & 31;
  const int wave = tid >> 5;
  const int wm = wave & 3;   // 0..3 -> 32 rows each
  const int wn = wave >> 2;  // 0..1 -> 64 cols each
  const int tileM = blockIdx.y * BM;
  const int tileN = blockIdx.x * BN;
  const int hi = lane >> 4, n16 = lane & 15;

  // Per-thread fill slots: 2 x 16B chunks for A, 2 for B (4096 bf16 per tile).
  auto fill = [&](int buf, int k0) {
#pragma unroll
    for (int i = 0; i < 2; ++i) {
      const int c = tid + i * 256;          // chunk id 0..511
      const int r = c >> 2;                 // row 0..127
      const int col = (c & 3) * 8;          // 0,8,16,24
      copy16_g2l(Ab + (long)(tileM + r) * K + k0 + col, &lA[buf][r * BK + col]);
      copy16_g2l(Bt + (long)(tileN + r) * K + k0 + col, &lB[buf][r * BK + col]);
    }
  };

  v8f acc[2][4] = {};

  fill(0, 0);
  const int ksteps = K / BK;
  int cur = 0;
  for (int kt = 0; kt < ksteps; ++kt) {
    wait_fills();
    __syncthreads();  // all waves' fills for `cur` visible; prev compute done
    if (kt + 1 < ksteps) fill(cur ^ 1, (kt + 1) * BK);
    if (kt + 2 < ksteps) {
      __builtin_prefetch(Ab + (long)(tileM + (tid >> 1)) * K + (kt + 2) * BK, 0, 1);
      __builtin_prefetch(Bt + (long)(tileN + (tid >> 1)) * K + (kt + 2) * BK, 0, 1);
    }

    const __bf16* a0 = &lA[cur][(wm * 32) * BK];
    const v16bf af0 = frag_a_16x32(a0, BK);
    const v16bf af1 = frag_a_16x32(a0 + 16 * BK, BK);
#pragma unroll
    for (int j = 0; j < 4; ++j) {
      const v16bf bf_ = frag_b_32x16(&lB[cur][(wn * 64 + j * 16) * BK], BK);
      acc[0][j] = wmma_bf16(af0, bf_, acc[0][j]);
      acc[1][j] = wmma_bf16(af1, bf_, acc[1][j]);
    }
    cur ^= 1;
  }

  // Epilogue. C layout: VGPR r -> M = r + 8*hi ; N = lane%16.
#pragma unroll
  for (int mi = 0; mi < 2; ++mi) {
#pragma unroll
    for (int j = 0; j < 4; ++j) {
#pragma unroll
      for (int r = 0; r < 8; ++r) {
        const int m = tileM + wm * 32 + mi * 16 + r + hi * 8;
        const int n = tileN + wn * 64 + j * 16 + n16;
        const float v = acc[mi][j][r] + bias[n];
        if (EPI == 0) {
          Cout[(long)m * N + n] = v;
        } else {
          const int which = n >> 10;  // 0=Q 1=K 2=V
          const int c = n & 1023;
          const int h = c >> 6, dd = c & 63;
          const int b = m >> 11, t = m & (Tn - 1);
          const long idx = (((long)(b * Hn + h)) * Tn + t) * DH + dd;
          if (which == 0)      Qb[idx] = (__bf16)(v * 0.125f);  // fold 1/sqrt(64)
          else if (which == 1) Kb[idx] = (__bf16)v;
          else                 Vb[idx] = (__bf16)v;
        }
      }
    }
  }
}

// ---------------------------------------------------------------------------
// Causal flash attention. One block per (b,h,64 query rows); 4 waves x 16 rows.
// Q frags resident; 32-key blocks; online softmax in f32; P->bf16 via LDS.
// Output written as bf16 [B,T,D] (feeds the projection GEMM's A operand).
// ---------------------------------------------------------------------------
__global__ __launch_bounds__(128) void attn_kernel(
    const __bf16* __restrict__ Qb, const __bf16* __restrict__ Kb,
    const __bf16* __restrict__ Vb, __bf16* __restrict__ Ob) {
  constexpr int BQ = 64, BKV = 32;
  __shared__ __bf16 lq[BQ * DH];      // [q][d]
  __shared__ __bf16 lk[BKV * DH];     // [key][d]
  __shared__ __bf16 lv[DH * BKV];     // [d][key]
  __shared__ __bf16 lp[4][16 * BKV];  // per-wave P tile [q][key]

  const int tid = threadIdx.x;
  const int lane = tid & 31;
  const int wave = tid >> 5;
  const int hi = lane >> 4, n16 = lane & 15;
  const int qblk = blockIdx.x & 31;  // T/BQ = 32
  const int bh = blockIdx.x >> 5;    // 0..B*H-1
  const int qBase = qblk * BQ;

  const __bf16* Qp = Qb + (long)bh * Tn * DH;
  const __bf16* Kp = Kb + (long)bh * Tn * DH;
  const __bf16* Vp = Vb + (long)bh * Tn * DH;

#pragma unroll
  for (int i = 0; i < (BQ * DH) / 128; ++i) {
    const int idx = tid + i * 128;
    lq[idx] = Qp[(long)qBase * DH + idx];
  }
  __syncthreads();
  const v16bf qa0 = frag_a_16x32(lq + (wave * 16) * DH + 0, DH);
  const v16bf qa1 = frag_a_16x32(lq + (wave * 16) * DH + 32, DH);

  float mrow[8], lrow[8];
  v8f acc[4] = {};
#pragma unroll
  for (int r = 0; r < 8; ++r) { mrow[r] = -1e30f; lrow[r] = 0.0f; }

  const int nkb = qBase / BKV + 2;  // causal: keys 0 .. qBase+63
  for (int kb = 0; kb < nkb; ++kb) {
    const int kBase = kb * BKV;
#pragma unroll
    for (int i = 0; i < (BKV * DH) / 128; ++i) {
      const int idx = tid + i * 128;
      const int key = idx >> 6, dd = idx & 63;
      lk[key * DH + dd]  = Kp[(long)(kBase + key) * DH + dd];
      lv[dd * BKV + key] = Vp[(long)(kBase + key) * DH + dd];
    }
    __syncthreads();

    // S = Q K^T
    v8f s0 = {}, s1 = {};
    s0 = wmma_bf16(qa0, frag_b_32x16(lk + 0, DH), s0);
    s0 = wmma_bf16(qa1, frag_b_32x16(lk + 32, DH), s0);
    s1 = wmma_bf16(qa0, frag_b_32x16(lk + 16 * DH, DH), s1);
    s1 = wmma_bf16(qa1, frag_b_32x16(lk + 16 * DH + 32, DH), s1);

    // Causal mask + online softmax. Row m = r + 8*hi; col n = lane%16.
    float alpha[8];
#pragma unroll
    for (int r = 0; r < 8; ++r) {
      const int mg = qBase + wave * 16 + r + hi * 8;
      const float v0 = (kBase + n16 > mg)      ? -1e30f : s0[r];
      const float v1 = (kBase + 16 + n16 > mg) ? -1e30f : s1[r];
      const float rmax = half_reduce_max(fmaxf(v0, v1));
      const float mnew = fmaxf(mrow[r], rmax);
      const float a = __expf(mrow[r] - mnew);
      mrow[r] = mnew;
      const float p0 = __expf(v0 - mnew);
      const float p1 = __expf(v1 - mnew);
      lrow[r] = lrow[r] * a + half_reduce_sum(p0 + p1);
      alpha[r] = a;
      lp[wave][(r + hi * 8) * BKV + n16]      = (__bf16)p0;
      lp[wave][(r + hi * 8) * BKV + 16 + n16] = (__bf16)p1;
    }
#pragma unroll
    for (int j = 0; j < 4; ++j)
#pragma unroll
      for (int r = 0; r < 8; ++r) acc[j][r] *= alpha[r];

    // O += P V
    const v16bf pf = frag_a_16x32(lp[wave], BKV);
#pragma unroll
    for (int j = 0; j < 4; ++j)
      acc[j] = wmma_bf16(pf, frag_b_32x16(lv + (j * 16) * BKV, BKV), acc[j]);

    __syncthreads();
  }

  const int b = bh >> 4, h = bh & 15;
#pragma unroll
  for (int j = 0; j < 4; ++j) {
#pragma unroll
    for (int r = 0; r < 8; ++r) {
      const int t = qBase + wave * 16 + r + hi * 8;
      const int col = h * DH + j * 16 + n16;
      Ob[((long)(b * Tn + t)) * Dm + col] = (__bf16)(acc[j][r] / lrow[r]);
    }
  }
}

// ---------------------------------------------------------------------------
// Host-side launch
// ---------------------------------------------------------------------------
extern "C" void kernel_launch(void* const* d_in, const int* in_sizes, int n_in,
                              void* d_out, int out_size, void* d_ws,
                              size_t ws_size, hipStream_t stream) {
  (void)in_sizes; (void)n_in; (void)out_size; (void)ws_size;
  const float* x      = (const float*)d_in[0];
  const float* w_attn = (const float*)d_in[1];
  const float* b_attn = (const float*)d_in[2];
  const float* w_proj = (const float*)d_in[3];
  const float* b_proj = (const float*)d_in[4];
  float* out = (float*)d_out;

  const long nX   = (long)Bn * Tn * Dm;      // 4M
  const long nWa  = (long)Dm * 3 * Dm;       // 3M
  const long nWp  = (long)Dm * Dm;           // 1M
  const long nQKV = (long)Bn * Hn * Tn * DH; // 4M

  __bf16* Xb  = (__bf16*)d_ws;
  __bf16* Wat = Xb + nX;    // w_attn^T  bf16 [3D][D]
  __bf16* Wpt = Wat + nWa;  // w_proj^T  bf16 [D][D]
  __bf16* Qb  = Wpt + nWp;
  __bf16* Kb  = Qb + nQKV;
  __bf16* Vb  = Kb + nQKV;
  __bf16* Ob  = Vb + nQKV;  // attention out bf16 [B,T,D]

  // 0) Stage operands in bf16 (weights transposed to [N][K]).
  cvt_f32_bf16_kernel<<<(int)(nX / 4 + 255) / 256, 256, 0, stream>>>(
      x, Xb, (int)(nX / 4));
  transpose_f32_bf16_kernel<<<dim3(3 * Dm / 32, Dm / 32), 256, 0, stream>>>(
      w_attn, Wat, Dm, 3 * Dm);
  transpose_f32_bf16_kernel<<<dim3(Dm / 32, Dm / 32), 256, 0, stream>>>(
      w_proj, Wpt, Dm, Dm);

  // 1) QKV GEMM -> bf16 Q(scaled)/K/V in [B,H,T,d]
  gemm_bf16_kernel<1><<<dim3(3 * Dm / 128, Bn * Tn / 128), 256, 0, stream>>>(
      Xb, Wat, b_attn, nullptr, Qb, Kb, Vb, Bn * Tn, 3 * Dm, Dm);

  // 2) Causal flash attention -> bf16 O [B,T,D]
  attn_kernel<<<dim3(Bn * Hn * (Tn / 64)), 128, 0, stream>>>(Qb, Kb, Vb, Ob);

  // 3) Output projection -> f32 out
  gemm_bf16_kernel<0><<<dim3(Dm / 128, Bn * Tn / 128), 256, 0, stream>>>(
      Ob, Wpt, b_proj, out, nullptr, nullptr, nullptr, Bn * Tn, Dm, Dm);
}
